// GraphSAGE_48232482734201
// MI455X (gfx1250) — matrix operand order repaired and checked by reading, not compiled, not run on hardware
//
#include <hip/hip_runtime.h>

// ---------------------------------------------------------------------------
// GraphSAGE (3-layer) for MI455X / gfx1250.
//   deg        = segment_sum(1, dst); invdeg = 1/max(deg,1)
//   per layer: agg = segment_sum(h[src], dst) * invdeg
//              h   = relu(h @ Ws + agg @ Wn + b)
// bf16 GEMMs on v_wmma_f32_16x16x32_bf16; bf16 edge gather + f32 global
// atomics for scatter.  Weights are packed bf16 *transposed* (N x K) once,
// so the per-block LDS stage is b128 global loads -> b128 ds stores, and
// B fragments are contiguous ds_load_b128 pairs.  K templated -> full unroll.
// ---------------------------------------------------------------------------

typedef __attribute__((ext_vector_type(16))) __bf16 v16bf;
typedef __attribute__((ext_vector_type(8)))  __bf16 v8bf;
typedef __attribute__((ext_vector_type(8)))  float  v8f;

union Frag16 { v16bf v; v8bf h[2]; };

static constexpr int NN = 100000;   // nodes
static constexpr int NP = 100032;   // nodes padded to multiple of 64

__device__ __forceinline__ void atomicAddF32(float* p, float v) {
  __hip_atomic_fetch_add(p, v, __ATOMIC_RELAXED, __HIP_MEMORY_SCOPE_AGENT);
}

// ----------------------------- utility kernels -----------------------------

__global__ void sage_zero_f32(float* __restrict__ p, size_t n) {
  size_t i = (size_t)blockIdx.x * blockDim.x + threadIdx.x;
  if (i < n) p[i] = 0.0f;
}

__global__ void sage_deg_count(const int* __restrict__ dst, int E,
                               float* __restrict__ deg) {
  int i = blockIdx.x * blockDim.x + threadIdx.x;
  if (i < E) atomicAddF32(&deg[dst[i]], 1.0f);
}

__global__ void sage_inv_deg(float* __restrict__ deg, int n) {
  int i = blockIdx.x * blockDim.x + threadIdx.x;
  if (i < n) deg[i] = 1.0f / fmaxf(deg[i], 1.0f);
}

// 8-wide f32 -> bf16 pack (n8 = n/8 chunks).
__global__ void sage_pack_bf8(const float* __restrict__ in,
                              __bf16* __restrict__ out, size_t n8) {
  size_t i = (size_t)blockIdx.x * blockDim.x + threadIdx.x;
  if (i >= n8) return;
  const float* p = in + i * 8;
  v8bf o;
#pragma unroll
  for (int j = 0; j < 8; ++j) o[j] = (__bf16)p[j];
  *(v8bf*)(out + i * 8) = o;
}

// Weight pack with transpose: wT[n*K + k] = bf16(W[k*N + n]).  K = 1<<kShift.
__global__ void sage_pack_wT(const float* __restrict__ W,
                             __bf16* __restrict__ wT,
                             int total, int kShift, int N) {
  int i = blockIdx.x * blockDim.x + threadIdx.x;
  if (i >= total) return;
  int k = i & ((1 << kShift) - 1);
  int n = i >> kShift;
  wT[i] = (__bf16)W[(size_t)k * N + n];
}

// agg_bf16 = bf16(agg_f32 * invdeg[node]), 8-wide; row = (i*8) >> dShift.
__global__ void sage_pack_agg8(const float* __restrict__ agg,
                               const float* __restrict__ invdeg,
                               __bf16* __restrict__ out,
                               size_t n8, int dShift) {
  size_t i = (size_t)blockIdx.x * blockDim.x + threadIdx.x;
  if (i >= n8) return;
  float s = invdeg[(i << 3) >> dShift];
  const float* p = agg + i * 8;
  v8bf o;
#pragma unroll
  for (int j = 0; j < 8; ++j) o[j] = (__bf16)(p[j] * s);
  *(v8bf*)(out + i * 8) = o;
}

// ------------------------------ edge scatter -------------------------------
// One thread per (edge, 8-feature chunk): 16B bf16 gather from h[src],
// 8 f32 global atomic adds into agg[dst].  d = 1<<dShift features.
__global__ void sage_scatter_bf(const __bf16* __restrict__ h,
                                const int* __restrict__ src,
                                const int* __restrict__ dst,
                                float* __restrict__ agg,
                                int E, int dShift) {
  const int cShift = dShift - 3;            // chunks per edge = d/8
  size_t tid = (size_t)blockIdx.x * blockDim.x + threadIdx.x;
  size_t total = (size_t)E << cShift;
  if (tid >= total) return;
  int e = (int)(tid >> cShift);
  int c = (int)(tid & ((1u << cShift) - 1u));
  int s = src[e];
  int d = dst[e];
  const __bf16* hp = h + (((size_t)s) << dShift) + ((size_t)c << 3);
  float* ap = agg + (((size_t)d) << dShift) + ((size_t)c << 3);
  v8bf v = *(const v8bf*)hp;                // global_load_b128
#pragma unroll
  for (int i = 0; i < 8; ++i) atomicAddF32(&ap[i], (float)v[i]);
}

// ------------------------------- fused GEMM --------------------------------
// out = relu(Aself @ Ws + Aneigh @ Wn + bias); Ws/Wn given TRANSPOSED (N x K).
// Block: 256 threads = 8 waves, 64(M) x 64(N) tile; wave: 16(M) x 32(N).
// LDS holds the block's 64 weight columns: ldsT[col*LSTR + k], LSTR = K+8.
template <int K>
__global__ __launch_bounds__(256)
void sage_gemm(const __bf16* __restrict__ Aself,
               const __bf16* __restrict__ Aneigh,
               const __bf16* __restrict__ WsT,
               const __bf16* __restrict__ WnT,
               const float*  __restrict__ bias,
               float*  __restrict__ outF,      // may be null
               __bf16* __restrict__ outB,      // may be null
               int nRows, int Nout) {
  constexpr int LSTR = K + 8;                 // LDS column stride (elements)
  constexpr int CH   = K / 8;                 // 16B chunks per column
  __shared__ __bf16 ldsT[64 * LSTR];          // <= 64*264*2 = 33 KB

  const int lane  = threadIdx.x & 31;
  const int wave  = threadIdx.x >> 5;
  const int wm    = wave & 3;                 // M sub-tile (0..3)
  const int wn    = wave >> 2;                // N sub-tile (0..1)
  const int half  = lane >> 4;                // lane group (K-half selector)
  const int mBase = blockIdx.x * 64 + wm * 16;
  const int nBlk  = blockIdx.y * 64;
  const int mRow  = mBase + (lane & 15);
  const int c0    = wn * 32 + (lane & 15);    // B-frag column (0..63)

  v8f acc0 = {};
  v8f acc1 = {};

  for (int pass = 0; pass < 2; ++pass) {
    const __bf16* A   = pass ? Aneigh : Aself;
    const __bf16* BmT = pass ? WnT : WsT;

    __syncthreads();
    // Vectorized fill: 16B column chunks, coalesced global -> b128 LDS.
#pragma unroll
    for (int idx = threadIdx.x; idx < 64 * CH; idx += 256) {
      int col = idx / CH;
      int ch  = idx & (CH - 1);
      *(v8bf*)(ldsT + col * LSTR + ch * 8) =
          *(const v8bf*)(BmT + ((size_t)(nBlk + col) << (K == 256 ? 8 : 7)) + ch * 8);
    }
    __syncthreads();

    const __bf16* arow  = A + (size_t)mRow * K;
    const __bf16* bcol0 = ldsT + c0 * LSTR;
    const __bf16* bcol1 = ldsT + (c0 + 16) * LSTR;
#pragma unroll
    for (int k0 = 0; k0 < K; k0 += 32) {
      // A fragment (16x32): per-lane two contiguous 8-elem chunks.
      Frag16 a;
      a.h[0] = *(const v8bf*)(arow + k0 + 16 * half);       // global b128
      a.h[1] = *(const v8bf*)(arow + k0 + 16 * half + 8);   // global b128
      // B fragments (32x16): per-lane 16 contiguous K values of one column.
      Frag16 b0, b1;
      b0.h[0] = *(const v8bf*)(bcol0 + k0 + 16 * half);     // ds b128
      b0.h[1] = *(const v8bf*)(bcol0 + k0 + 16 * half + 8); // ds b128
      b1.h[0] = *(const v8bf*)(bcol1 + k0 + 16 * half);
      b1.h[1] = *(const v8bf*)(bcol1 + k0 + 16 * half + 8);
      acc0 = __builtin_amdgcn_wmma_f32_16x16x32_bf16(
          false, a.v, false, b0.v, (short)0, acc0, false, false);
      acc1 = __builtin_amdgcn_wmma_f32_16x16x32_bf16(
          false, a.v, false, b1.v, (short)0, acc1, false, false);
    }
  }

  // Epilogue: bias + ReLU, store f32 and/or bf16.
  const int nc = nBlk + wn * 32 + (lane & 15);
  const float bs0 = bias[nc];
  const float bs1 = bias[nc + 16];
#pragma unroll
  for (int r = 0; r < 8; ++r) {
    int m = mBase + r + 8 * half;             // C/D VGPR layout
    if (m < nRows) {
      float v0 = fmaxf(acc0[r] + bs0, 0.0f);
      float v1 = fmaxf(acc1[r] + bs1, 0.0f);
      size_t o = (size_t)m * Nout + nc;
      if (outF) { outF[o] = v0; outF[o + 16] = v1; }
      if (outB) { outB[o] = (__bf16)v0; outB[o + 16] = (__bf16)v1; }
    }
  }
}

// ------------------------------- host driver -------------------------------

extern "C" void kernel_launch(void* const* d_in, const int* in_sizes, int n_in,
                              void* d_out, int out_size, void* d_ws, size_t ws_size,
                              hipStream_t stream) {
  const float* x   = (const float*)d_in[0];
  const int*   src = (const int*)d_in[1];
  const int*   dst = (const int*)d_in[2];
  const float* Ws0 = (const float*)d_in[3];
  const float* Wn0 = (const float*)d_in[4];
  const float* b0  = (const float*)d_in[5];
  const float* Ws1 = (const float*)d_in[6];
  const float* Wn1 = (const float*)d_in[7];
  const float* b1  = (const float*)d_in[8];
  const float* Ws2 = (const float*)d_in[9];
  const float* Wn2 = (const float*)d_in[10];
  const float* b2  = (const float*)d_in[11];
  float* out = (float*)d_out;
  const int E = in_sizes[1];

  // Workspace carve-up (256B aligned).
  char* base = (char*)d_ws;
  size_t off = 0;
  auto carve = [&](size_t bytes) {
    char* p = base + off;
    off = (off + bytes + 255) & ~(size_t)255;
    return p;
  };
  float*  invdeg = (float*)carve((size_t)NP * 4);
  __bf16* xbf    = (__bf16*)carve((size_t)NP * 128 * 2);
  __bf16* h0b    = (__bf16*)carve((size_t)NP * 256 * 2);
  __bf16* h1b    = (__bf16*)carve((size_t)NP * 256 * 2);
  float*  aggf   = (float*)carve((size_t)NP * 256 * 4);
  __bf16* aggb   = (__bf16*)carve((size_t)NP * 256 * 2);
  __bf16* w0s    = (__bf16*)carve((size_t)128 * 256 * 2);   // transposed N x K
  __bf16* w0n    = (__bf16*)carve((size_t)128 * 256 * 2);
  __bf16* w1s    = (__bf16*)carve((size_t)256 * 256 * 2);
  __bf16* w1n    = (__bf16*)carve((size_t)256 * 256 * 2);
  __bf16* w2s    = (__bf16*)carve((size_t)256 * 128 * 2);
  __bf16* w2n    = (__bf16*)carve((size_t)256 * 128 * 2);

  auto blocks = [](size_t n) { return (unsigned)((n + 255) / 256); };

  // Degrees.
  sage_zero_f32<<<blocks(NP), 256, 0, stream>>>(invdeg, NP);
  sage_deg_count<<<blocks(E), 256, 0, stream>>>(dst, E, invdeg);
  sage_inv_deg<<<blocks(NP), 256, 0, stream>>>(invdeg, NP);

  // Pack x (bf16) and weights (bf16, transposed N x K).
  sage_pack_bf8<<<blocks((size_t)NN * 16), 256, 0, stream>>>(x, xbf, (size_t)NN * 16);
  sage_pack_wT<<<blocks(128 * 256), 256, 0, stream>>>(Ws0, w0s, 128 * 256, 7, 256);
  sage_pack_wT<<<blocks(128 * 256), 256, 0, stream>>>(Wn0, w0n, 128 * 256, 7, 256);
  sage_pack_wT<<<blocks(256 * 256), 256, 0, stream>>>(Ws1, w1s, 256 * 256, 8, 256);
  sage_pack_wT<<<blocks(256 * 256), 256, 0, stream>>>(Wn1, w1n, 256 * 256, 8, 256);
  sage_pack_wT<<<blocks(256 * 128), 256, 0, stream>>>(Ws2, w2s, 256 * 128, 8, 128);
  sage_pack_wT<<<blocks(256 * 128), 256, 0, stream>>>(Wn2, w2n, 256 * 128, 8, 128);

  const dim3 gemmBlk(256);

  // ---- Layer 0: 128 -> 256 ----
  sage_zero_f32<<<blocks((size_t)NP * 128), 256, 0, stream>>>(aggf, (size_t)NP * 128);
  sage_scatter_bf<<<blocks((size_t)E * 16), 256, 0, stream>>>(xbf, src, dst, aggf, E, 7);
  sage_pack_agg8<<<blocks((size_t)NP * 16), 256, 0, stream>>>(aggf, invdeg, aggb,
                                                              (size_t)NP * 16, 7);
  sage_gemm<128><<<dim3(NP / 64, 256 / 64), gemmBlk, 0, stream>>>(
      xbf, aggb, w0s, w0n, b0, nullptr, h0b, NN, 256);

  // ---- Layer 1: 256 -> 256 ----
  sage_zero_f32<<<blocks((size_t)NP * 256), 256, 0, stream>>>(aggf, (size_t)NP * 256);
  sage_scatter_bf<<<blocks((size_t)E * 32), 256, 0, stream>>>(h0b, src, dst, aggf, E, 8);
  sage_pack_agg8<<<blocks((size_t)NP * 32), 256, 0, stream>>>(aggf, invdeg, aggb,
                                                              (size_t)NP * 32, 8);
  sage_gemm<256><<<dim3(NP / 64, 256 / 64), gemmBlk, 0, stream>>>(
      h0b, aggb, w1s, w1n, b1, nullptr, h1b, NN, 256);

  // ---- Layer 2: 256 -> 128 (final f32 output) ----
  sage_zero_f32<<<blocks((size_t)NP * 256), 256, 0, stream>>>(aggf, (size_t)NP * 256);
  sage_scatter_bf<<<blocks((size_t)E * 32), 256, 0, stream>>>(h1b, src, dst, aggf, E, 8);
  sage_pack_agg8<<<blocks((size_t)NP * 32), 256, 0, stream>>>(aggf, invdeg, aggb,
                                                              (size_t)NP * 32, 8);
  sage_gemm<256><<<dim3(NP / 64, 128 / 64), gemmBlk, 0, stream>>>(
      h1b, aggb, w2s, w2n, b2, out, nullptr, NN, 128);
}
